// TreeConnect_69827578298457
// MI455X (gfx1250) — matrix operand order repaired and checked by reading, not compiled, not run on hardware
//
#include <hip/hip_runtime.h>

typedef __attribute__((ext_vector_type(16))) _Float16 v16h;
typedef __attribute__((ext_vector_type(8)))  float    v8f;
typedef __attribute__((ext_vector_type(4)))  float    v4f;

#define BATCH 4096
#define D     4096
#define SEC   256
#define NBLK  16

// ---------------------------------------------------------------------------
// Pack W1 block-diagonal blocks to f16: W1c[(g*256+n)*256 + k] = W1[g*256+n][g*256+k]
// ---------------------------------------------------------------------------
__global__ __launch_bounds__(256) void pack_w1_kernel(const float* __restrict__ W1,
                                                      _Float16* __restrict__ W1c) {
    int idx = blockIdx.x * 256 + threadIdx.x;      // 0 .. 4096*256-1
    int o = idx >> 8;                              // global output row
    int i = idx & 255;                             // local input col
    int g = o >> 8;                                // block
    W1c[idx] = (_Float16)W1[(size_t)o * D + (size_t)g * SEC + i];
}

// ---------------------------------------------------------------------------
// Pack per-residue 16x16 tables for layers 2 and 3, layout [i][k][r] (coalesced in r)
//   W2c: lower-triangular (k<=i) taps of m2;  W3c: full 16 taps of m3
// ---------------------------------------------------------------------------
__global__ __launch_bounds__(256) void pack_w23_kernel(const float* __restrict__ W2,
                                                       const float* __restrict__ W3,
                                                       float* __restrict__ W2c,
                                                       float* __restrict__ W3c) {
    int idx = blockIdx.x * 256 + threadIdx.x;      // 0 .. 65535 ; idx = (i*16+k)*256 + r
    int r = idx & 255;
    int k = (idx >> 8) & 15;
    int i = idx >> 12;
    size_t src = (size_t)(i * SEC + r) * D + (size_t)k * SEC + r;
    W2c[idx] = (k <= i) ? W2[src] : 0.0f;
    W3c[idx] = W3[src];
}

// ---------------------------------------------------------------------------
// Layer 1: 16 block GEMMs (4096x256) x (256x256)^T via v_wmma_f32_16x16x32_f16
// One wave = 16 rows x 64 cols (4 WMMA accumulators), K loop of 32.
// ---------------------------------------------------------------------------
__global__ __launch_bounds__(256) void layer1_wmma_kernel(const float* __restrict__ x,
                                                          const _Float16* __restrict__ W1c,
                                                          const float* __restrict__ b1,
                                                          float* __restrict__ h1) {
    const int lane = threadIdx.x & 31;
    const int wave = threadIdx.x >> 5;
    const int job  = blockIdx.x * 8 + wave;        // 16 * 256 * 4 = 16384 wave jobs
    const int g    = job >> 10;                    // weight block 0..15
    const int mt   = (job >> 2) & 255;             // M tile 0..255
    const int ns   = job & 3;                      // N strip 0..3 (64 cols each)
    const int m0   = mt << 4;
    const int n0   = ns << 6;

    const int lrow = lane & 15;
    const int hi   = lane >> 4;                    // half-wave select
    const int kA   = hi ? 16 : 0;                  // A: K base, contiguous k = 16*hi + s
    const int kB   = hi ? 16 : 0;                  // B: K base, contiguous k = 16*hi + s (matches A)

    const float*    xp = x + (size_t)(m0 + lrow) * D + (size_t)g * SEC;
    const _Float16* wb = W1c + ((size_t)g << 16) + (size_t)(n0 + lrow) * SEC + kB;

    v8f acc0 = {}, acc1 = {}, acc2 = {}, acc3 = {};

    for (int kk = 0; kk < SEC; kk += 32) {
        // ---- A tile: x row in f32 -> f16, contiguous k = 16*hi + s per lane
        v4f a0 = *(const v4f*)(xp + kk + kA);
        v4f a1 = *(const v4f*)(xp + kk + kA + 4);
        v4f a2 = *(const v4f*)(xp + kk + kA + 8);
        v4f a3 = *(const v4f*)(xp + kk + kA + 12);
        v16h a;
        a[0]  = (_Float16)a0[0]; a[1]  = (_Float16)a0[1];
        a[2]  = (_Float16)a0[2]; a[3]  = (_Float16)a0[3];
        a[4]  = (_Float16)a1[0]; a[5]  = (_Float16)a1[1];
        a[6]  = (_Float16)a1[2]; a[7]  = (_Float16)a1[3];
        a[8]  = (_Float16)a2[0]; a[9]  = (_Float16)a2[1];
        a[10] = (_Float16)a2[2]; a[11] = (_Float16)a2[3];
        a[12] = (_Float16)a3[0]; a[13] = (_Float16)a3[1];
        a[14] = (_Float16)a3[2]; a[15] = (_Float16)a3[3];

        // ---- B tiles: packed f16 weights, 32B contiguous per lane
        v16h bv0 = *(const v16h*)(wb + (size_t)( 0) * SEC + kk);
        v16h bv1 = *(const v16h*)(wb + (size_t)(16) * SEC + kk);
        v16h bv2 = *(const v16h*)(wb + (size_t)(32) * SEC + kk);
        v16h bv3 = *(const v16h*)(wb + (size_t)(48) * SEC + kk);

        acc0 = __builtin_amdgcn_wmma_f32_16x16x32_f16(false, a, false, bv0, (short)0, acc0, false, false);
        acc1 = __builtin_amdgcn_wmma_f32_16x16x32_f16(false, a, false, bv1, (short)0, acc1, false, false);
        acc2 = __builtin_amdgcn_wmma_f32_16x16x32_f16(false, a, false, bv2, (short)0, acc2, false, false);
        acc3 = __builtin_amdgcn_wmma_f32_16x16x32_f16(false, a, false, bv3, (short)0, acc3, false, false);
    }

    // ---- epilogue: bias + ReLU, C/D layout: VGPR r, lane -> row m0 + r + hi*8, col lane&15
    float* hbase = h1 + (size_t)g * SEC;
    v8f accs[4] = {acc0, acc1, acc2, acc3};
#pragma unroll
    for (int nt = 0; nt < 4; ++nt) {
        const int n   = n0 + nt * 16 + lrow;
        const float bias = b1[g * SEC + n];
        v8f acc = accs[nt];
#pragma unroll
        for (int rr = 0; rr < 8; ++rr) {
            float v = acc[rr] + bias;
            v = v > 0.0f ? v : 0.0f;
            hbase[(size_t)(m0 + rr + hi * 8) * D + n] = v;
        }
    }
}

// ---------------------------------------------------------------------------
// Fused layers 2+3: per (batch row b, residue r): h1[b, k*256+r] -> out[b, m*256+r]
//   u[i] = relu(b2 + sum_{k<=i} W2c[i][k][r] * v[k])
//   out[m] = b3 + sum_k W3c[m][k][r] * u[k]
// All global accesses coalesced in r; weight tables (512 KB) are L2-resident.
// ---------------------------------------------------------------------------
__global__ __launch_bounds__(256) void layer23_kernel(const float* __restrict__ h1,
                                                      const float* __restrict__ W2c,
                                                      const float* __restrict__ b2,
                                                      const float* __restrict__ W3c,
                                                      const float* __restrict__ b3,
                                                      float* __restrict__ out) {
    const int b = blockIdx.x;
    const int r = threadIdx.x;
    const float* hp = h1 + (size_t)b * D + r;

    float v[16];
#pragma unroll
    for (int k = 0; k < 16; ++k) v[k] = hp[k << 8];

    float u[16];
#pragma unroll
    for (int i = 0; i < 16; ++i) {
        float acc = b2[(i << 8) + r];
#pragma unroll
        for (int k = 0; k < 16; ++k) {
            if (k <= i) acc += W2c[((i * 16 + k) << 8) + r] * v[k];
        }
        u[i] = acc > 0.0f ? acc : 0.0f;
    }

    float* op = out + (size_t)b * D + r;
#pragma unroll
    for (int m = 0; m < 16; ++m) {
        float acc = b3[(m << 8) + r];
#pragma unroll
        for (int k = 0; k < 16; ++k)
            acc += W3c[((m * 16 + k) << 8) + r] * u[k];
        op[m << 8] = acc;
    }
}

// ---------------------------------------------------------------------------
extern "C" void kernel_launch(void* const* d_in, const int* in_sizes, int n_in,
                              void* d_out, int out_size, void* d_ws, size_t ws_size,
                              hipStream_t stream) {
    const float* x  = (const float*)d_in[0];
    const float* W1 = (const float*)d_in[1];
    const float* b1 = (const float*)d_in[2];
    const float* W2 = (const float*)d_in[3];
    const float* b2 = (const float*)d_in[4];
    const float* W3 = (const float*)d_in[5];
    const float* b3 = (const float*)d_in[6];
    // d_in[7..9] = m1,m2,m3 dense masks: structure is known analytically, not read.
    float* out = (float*)d_out;

    char* ws = (char*)d_ws;
    float*    h1  = (float*)ws;                                   // 64 MB
    _Float16* W1c = (_Float16*)(ws + ((size_t)64 << 20));         //  2 MB
    float*    W2c = (float*)   (ws + ((size_t)66 << 20));         // 256 KB
    float*    W3c = (float*)   (ws + ((size_t)66 << 20) + (256u << 10)); // 256 KB

    pack_w1_kernel <<<4096, 256, 0, stream>>>(W1, W1c);
    pack_w23_kernel<<< 256, 256, 0, stream>>>(W2, W3, W2c, W3c);
    layer1_wmma_kernel<<<2048, 256, 0, stream>>>(x, W1c, b1, h1);
    layer23_kernel<<<BATCH, 256, 0, stream>>>(h1, W2c, b2, W3c, b3, out);
}
